// Attenion_operation_87548613362191
// MI455X (gfx1250) — matrix operand order repaired and checked by reading, not compiled, not run on hardware
//
#include <hip/hip_runtime.h>
#include <hip/hip_bf16.h>

// Transformer encoder layer, MI455X/gfx1250, wave32 + WMMA bf16 (K=32).
// Flash-style attention with async global->LDS double-buffered K/V staging.
// Weights pre-packed into B-fragment layout (one contiguous v16bf per lane).

#define LNUM   8
#define SEQ    2048
#define DMODEL 128
#define NHEAD  4
#define HDIM   32
#define FDIM   384
#define NTOK   (LNUM * SEQ)      // 16384 tokens
#define NEGB   (-1.0e9f)
#define LN_EPS 1e-5f

typedef __attribute__((ext_vector_type(16))) __bf16 v16bf;
typedef __attribute__((ext_vector_type(8)))  __bf16 v8bf;
typedef __attribute__((ext_vector_type(4)))  __bf16 v4bf;
typedef __attribute__((ext_vector_type(8)))  float  v8f;
typedef __attribute__((ext_vector_type(4)))  float  v4f;

#if defined(__has_builtin)
#if __has_builtin(__builtin_amdgcn_sched_barrier)
#define SCHED_BARRIER() __builtin_amdgcn_sched_barrier(0)
#endif
#endif
#ifndef SCHED_BARRIER
#define SCHED_BARRIER() asm volatile("" ::: "memory")
#endif

__device__ __forceinline__ v8f wmma_bf16(v16bf a, v16bf b, v8f c) {
  return __builtin_amdgcn_wmma_f32_16x16x32_bf16(false, a, false, b, (short)0, c,
                                                 false, false);
}

// Async global -> LDS copy of 16 bytes (tracked by ASYNCcnt), gfx1250 path.
__device__ __forceinline__ void async_g2l_b128(unsigned ldsAddr, const void* gaddr) {
  asm volatile("global_load_async_to_lds_b128 %0, %1, off"
               :: "v"(ldsAddr), "v"(gaddr) : "memory");
}
__device__ __forceinline__ void wait_async0() {
  asm volatile("s_wait_asynccnt 0x0" ::: "memory");
}

// A-operand (16x32 bf16, row-major source). lanes 0-15 hold rows
// 0-15 with K={0..15}; lanes 16-31 hold same rows with K={16..31}.
__device__ __forceinline__ v16bf load_a_frag(const __bf16* A, int lda, long row,
                                             int kbase) {
  const int lane = threadIdx.x & 31;
  const int grp  = (lane >> 4) & 1;
  const __bf16* p = A + row * (long)lda + kbase + grp * 16;
  v8bf c0 = *(const v8bf*)(p);
  v8bf c1 = *(const v8bf*)(p + 8);
  v16bf r;
  #pragma unroll
  for (int i = 0; i < 8; ++i) { r[i] = c0[i]; r[i + 8] = c1[i]; }
  return r;
}

// Packed B: layout [kb][nt][lane][16] bf16 -> one 32B contiguous load per lane.
__device__ __forceinline__ v16bf load_b_packed(const __bf16* P, int ntiles,
                                               int kb, int nt) {
  const int lane = threadIdx.x & 31;
  return *(const v16bf*)(P + (((long)kb * ntiles + nt) * 32 + lane) * 16);
}

// ---------------------------------------------------------------- kernels ---

__global__ void k_cvt_bf16(const float* __restrict__ src, __bf16* __restrict__ dst,
                           int n4) {            // n4 = n/4
  for (int i = blockIdx.x * blockDim.x + threadIdx.x; i < n4;
       i += gridDim.x * blockDim.x) {
    v4f f = *(const v4f*)(src + (long)i * 4);
    v4bf o;
    #pragma unroll
    for (int e = 0; e < 4; ++e) o[e] = (__bf16)f[e];
    *(v4bf*)(dst + (long)i * 4) = o;
  }
}

// Pack f32 W[K x N] into bf16 B-fragment layout [kb][nt][lane][16].
__global__ void k_pack_b(const float* __restrict__ W, __bf16* __restrict__ P,
                         int K, int N) {
  const int total = (K / 32) * (N / 16) * 32;
  const int idx = blockIdx.x * blockDim.x + threadIdx.x;
  if (idx >= total) return;
  const int lane = idx & 31;
  const int rest = idx >> 5;
  const int ntiles = N / 16;
  const int nt = rest % ntiles;
  const int kb = rest / ntiles;
  const int grp = lane >> 4, n15 = lane & 15;
  const float* src = W + (long)(kb * 32 + grp * 16) * N + nt * 16 + n15;
  v16bf r;
  #pragma unroll
  for (int e = 0; e < 16; ++e) r[e] = (__bf16)src[(long)e * N];
  *(v16bf*)(P + (long)idx * 16) = r;
}

// qkv = xb @ qkv_w ; 32-row M tile per block (two A sets share each B frag);
// scatter into q[l,h,s,hd], k[l,h,s,hd], v^T[l,h,hd,s] (bf16)
__global__ void k_qkv(const __bf16* __restrict__ xb, const __bf16* __restrict__ pW,
                      __bf16* __restrict__ qb, __bf16* __restrict__ kbuf,
                      __bf16* __restrict__ vtb) {
  const int rt   = blockIdx.x;               // 32-row tile over NTOK
  const int wave = threadIdx.x >> 5;
  const int lane = threadIdx.x & 31;
  const int grp  = (lane >> 4) & 1;
  const int n15  = lane & 15;
  const long rowA = (long)rt * 32 + n15;
  const int  l     = rt >> 6;                // block-uniform (64 tiles per l)
  const int  sbase = (rt & 63) * 32;

  v16bf a0[4], a1[4];
  #pragma unroll
  for (int k4 = 0; k4 < 4; ++k4) {
    a0[k4] = load_a_frag(xb, DMODEL, rowA,      k4 * 32);
    a1[k4] = load_a_frag(xb, DMODEL, rowA + 16, k4 * 32);
  }

  for (int ct0 = wave; ct0 < 24; ct0 += 4) {
    const int ct = __builtin_amdgcn_readfirstlane(ct0);   // wave-uniform
    __builtin_prefetch(pW + ((long)((ct + 4) % 24) * 32) * 16, 0, 1);
    v16bf b[4];
    #pragma unroll
    for (int k4 = 0; k4 < 4; ++k4) b[k4] = load_b_packed(pW, 24, k4, ct);
    SCHED_BARRIER();                         // keep loads ahead of WMMAs
    v8f acc0 = {}, acc1 = {};
    #pragma unroll
    for (int k4 = 0; k4 < 4; ++k4) {         // two independent WMMA chains
      acc0 = wmma_bf16(a0[k4], b[k4], acc0);
      acc1 = wmma_bf16(a1[k4], b[k4], acc1);
    }

    const int sec  = ct >> 3;                // 0=q 1=k 2=v (scalar branches)
    const int head = (ct & 7) >> 1;
    const int hdid = ((ct & 1) << 4) + n15;
    const long hbase = (long)(l * NHEAD + head);
    #pragma unroll
    for (int r = 0; r < 8; ++r) {
      const int s0 = sbase + r + grp * 8;
      const int s1 = s0 + 16;
      const __bf16 v0 = (__bf16)acc0[r];
      const __bf16 v1 = (__bf16)acc1[r];
      if (sec == 0) {
        qb[(hbase * SEQ + s0) * HDIM + hdid] = v0;
        qb[(hbase * SEQ + s1) * HDIM + hdid] = v1;
      } else if (sec == 1) {
        kbuf[(hbase * SEQ + s0) * HDIM + hdid] = v0;
        kbuf[(hbase * SEQ + s1) * HDIM + hdid] = v1;
      } else {
        vtb[(hbase * HDIM + hdid) * SEQ + s0] = v0;
        vtb[(hbase * HDIM + hdid) * SEQ + s1] = v1;
      }
    }
  }
}

// Flash attention: 4 waves per block = 64 queries; K/V 32-key chunks shared
// through LDS, double-buffered via async global->LDS loads (ASYNCcnt).
__global__ void k_attn(const __bf16* __restrict__ qb, const __bf16* __restrict__ kbuf,
                       const __bf16* __restrict__ vtb, const int* __restrict__ mask,
                       __bf16* __restrict__ ctxb) {
  __shared__ __bf16 kT[2][32 * 32];   // [buf][key][hd]
  __shared__ __bf16 vT[2][32 * 32];   // [buf][hd][key]
  __shared__ __bf16 pt[4][16 * 32];   // per-wave P tile
  __shared__ float  cmask[SEQ];       // column mask as float
  const int bid  = blockIdx.x;        // L*H*(SEQ/64)
  const int qt4  = bid & 31;
  const int h    = (bid >> 5) & 3;
  const int l    = bid >> 7;
  const int wave = threadIdx.x >> 5;
  const int lane = threadIdx.x & 31;
  const int grp  = (lane >> 4) & 1;
  const int n15  = lane & 15;
  const long base  = (long)(l * NHEAD + h) * SEQ;
  const int  qbase = qt4 * 64 + wave * 16;
  const float scale = 0.17677669529663687f;  // 1/sqrt(32)

  // one-time column-mask preload: 128 threads x 16 entries
  #pragma unroll
  for (int i = 0; i < 16; ++i) {
    const int idx = threadIdx.x * 16 + i;
    cmask[idx] = (float)mask[l * SEQ + idx];
  }

  // cooperative staging: 128 threads x 16B cover one 32x32 bf16 chunk each
  const int srow = threadIdx.x >> 2;          // 0..31
  const int sseg = (threadIdx.x & 3) * 8;     // bf16 offset, 16B granules
  const __bf16* kgBase = kbuf + base * HDIM;
  const __bf16* vgBase = vtb + base * HDIM;   // == (l*NHEAD+h)*HDIM*SEQ

  v16bf aq = load_a_frag(qb + base * HDIM, HDIM, qbase + n15, 0);

  v8f o0 = {}, o1 = {};
  float mi[8], li[8];
  #pragma unroll
  for (int r = 0; r < 8; ++r) { mi[r] = -3.0e38f; li[r] = 0.0f; }

  // prologue: stage chunk 0 into buffer 0
  async_g2l_b128((unsigned)(size_t)&kT[0][srow * 32 + sseg],
                 kgBase + (long)srow * HDIM + sseg);
  async_g2l_b128((unsigned)(size_t)&vT[0][srow * 32 + sseg],
                 vgBase + (long)srow * SEQ + sseg);
  wait_async0();
  __syncthreads();

  float mrowv[8];
  #pragma unroll
  for (int r = 0; r < 8; ++r) mrowv[r] = cmask[qbase + r + grp * 8];

  int buf = 0;
  for (int kt = 0; kt < SEQ; kt += 32) {
    if (kt + 32 < SEQ) {   // stage next chunk into the other buffer (async)
      const int nb = buf ^ 1;
      async_g2l_b128((unsigned)(size_t)&kT[nb][srow * 32 + sseg],
                     kgBase + ((long)kt + 32 + srow) * HDIM + sseg);
      async_g2l_b128((unsigned)(size_t)&vT[nb][srow * 32 + sseg],
                     vgBase + (long)srow * SEQ + kt + 32 + sseg);
    }

    // both score WMMAs issued back-to-back (independent accumulators)
    v16bf bk0 = *(const v16bf*)(&kT[buf][(n15)      * 32 + grp * 16]);
    v16bf bk1 = *(const v16bf*)(&kT[buf][(16 + n15) * 32 + grp * 16]);
    v8f s0 = {}, s1 = {};
    s0 = wmma_bf16(aq, bk0, s0);
    s1 = wmma_bf16(aq, bk1, s1);
    // V fragments loaded early; softmax VALU co-executes with matrix pipe
    v16bf bv0 = *(const v16bf*)(&vT[buf][(n15)      * 32 + grp * 16]);
    v16bf bv1 = *(const v16bf*)(&vT[buf][(16 + n15) * 32 + grp * 16]);
    const float mcol0 = cmask[kt + n15];
    const float mcol1 = cmask[kt + 16 + n15];
    SCHED_BARRIER();       // pin WMMA + V loads ahead of the softmax VALU

    float sc[2][8];
    #pragma unroll
    for (int r = 0; r < 8; ++r) {
      const float am0 = mrowv[r] * mcol0;
      const float am1 = mrowv[r] * mcol1;
      sc[0][r] = am0 * (s0[r] * scale) + (am0 - 1.0f) * NEGB;
      sc[1][r] = am1 * (s1[r] * scale) + (am1 - 1.0f) * NEGB;
    }
    // online softmax: row r spans the 16-lane group
    #pragma unroll
    for (int r = 0; r < 8; ++r) {
      float mx = fmaxf(sc[0][r], sc[1][r]);
      #pragma unroll
      for (int off = 1; off < 16; off <<= 1) mx = fmaxf(mx, __shfl_xor(mx, off, 16));
      const float nm   = fmaxf(mi[r], mx);
      const float corr = __expf(mi[r] - nm);
      const float p0 = __expf(sc[0][r] - nm);
      const float p1 = __expf(sc[1][r] - nm);
      float ps = p0 + p1;
      #pragma unroll
      for (int off = 1; off < 16; off <<= 1) ps += __shfl_xor(ps, off, 16);
      li[r] = li[r] * corr + ps;
      mi[r] = nm;
      o0[r] *= corr; o1[r] *= corr;
      pt[wave][(r + grp * 8) * 32 + n15]      = (__bf16)p0;
      pt[wave][(r + grp * 8) * 32 + 16 + n15] = (__bf16)p1;
    }
    // per-wave DS ops are in-order; compiler barrier keeps store->load order
    asm volatile("" ::: "memory");
    v16bf ap = load_a_frag(pt[wave], 32, n15, 0);     // P as A-operand (16x32)
    o0 = wmma_bf16(ap, bv0, o0);
    o1 = wmma_bf16(ap, bv1, o1);

    wait_async0();        // next chunk landed
    __syncthreads();      // all waves done with 'buf' before it is reused
    buf ^= 1;
  }
  #pragma unroll
  for (int r = 0; r < 8; ++r) {
    const float rl = 1.0f / li[r];
    const long  t  = (long)l * SEQ + qbase + r + grp * 8;
    ctxb[t * DMODEL + h * HDIM + n15]      = (__bf16)(o0[r] * rl);
    ctxb[t * DMODEL + h * HDIM + 16 + n15] = (__bf16)(o1[r] * rl);
  }
}

// attn_out = ctx @ out_w ; + x residual ; LayerNorm(g1,be1) -> h (f32) + hb (bf16)
__global__ void k_proj_ln1(const __bf16* __restrict__ ctxb, const __bf16* __restrict__ pW,
                           const float* __restrict__ x, const float* __restrict__ g1,
                           const float* __restrict__ be1, float* __restrict__ hres,
                           __bf16* __restrict__ hb) {
  __shared__ float tile[16 * DMODEL];
  const int rt   = blockIdx.x;
  const int wave = threadIdx.x >> 5;
  const int lane = threadIdx.x & 31;
  const int grp  = (lane >> 4) & 1;
  const int n15  = lane & 15;
  const long rowA = (long)rt * 16 + n15;

  v16bf a[4];
  #pragma unroll
  for (int k4 = 0; k4 < 4; ++k4) a[k4] = load_a_frag(ctxb, DMODEL, rowA, k4 * 32);

  for (int nt = wave; nt < 8; nt += 4) {
    v16bf b[4];
    #pragma unroll
    for (int k4 = 0; k4 < 4; ++k4) b[k4] = load_b_packed(pW, 8, k4, nt);
    SCHED_BARRIER();
    v8f acc = {};
    #pragma unroll
    for (int k4 = 0; k4 < 4; ++k4) acc = wmma_bf16(a[k4], b[k4], acc);
    const int col = nt * 16 + n15;
    #pragma unroll
    for (int r = 0; r < 8; ++r) {
      const int  rr = r + grp * 8;
      const long t  = (long)rt * 16 + rr;
      tile[rr * DMODEL + col] = acc[r] + x[t * DMODEL + col];
    }
  }
  __syncthreads();
  if (threadIdx.x < 16) {
    const int  rr = threadIdx.x;
    const long t  = (long)rt * 16 + rr;
    float mean = 0.0f;
    for (int c = 0; c < DMODEL; ++c) mean += tile[rr * DMODEL + c];
    mean *= (1.0f / DMODEL);
    float var = 0.0f;
    for (int c = 0; c < DMODEL; ++c) {
      const float d = tile[rr * DMODEL + c] - mean; var += d * d;
    }
    var *= (1.0f / DMODEL);
    const float rstd = rsqrtf(var + LN_EPS);
    for (int c = 0; c < DMODEL; ++c) {
      const float hv = g1[c] * (tile[rr * DMODEL + c] - mean) * rstd + be1[c];
      hres[t * DMODEL + c] = hv;
      hb[t * DMODEL + c]   = (__bf16)hv;
    }
  }
}

// ff1 = relu(hb @ w1 + b1) -> bf16 ; 32-row M tile per block
__global__ void k_ffn1(const __bf16* __restrict__ hb, const __bf16* __restrict__ pW,
                       const float* __restrict__ b1, __bf16* __restrict__ ff1b) {
  const int rt   = blockIdx.x;               // 32-row tile
  const int wave = threadIdx.x >> 5;
  const int lane = threadIdx.x & 31;
  const int grp  = (lane >> 4) & 1;
  const int n15  = lane & 15;
  const long rowA = (long)rt * 32 + n15;

  v16bf a0[4], a1[4];
  #pragma unroll
  for (int k4 = 0; k4 < 4; ++k4) {
    a0[k4] = load_a_frag(hb, DMODEL, rowA,      k4 * 32);
    a1[k4] = load_a_frag(hb, DMODEL, rowA + 16, k4 * 32);
  }

  for (int ct0 = wave; ct0 < 24; ct0 += 4) {
    const int ct = __builtin_amdgcn_readfirstlane(ct0);
    __builtin_prefetch(pW + ((long)((ct + 4) % 24) * 32) * 16, 0, 1);
    v16bf b[4];
    #pragma unroll
    for (int k4 = 0; k4 < 4; ++k4) b[k4] = load_b_packed(pW, 24, k4, ct);
    SCHED_BARRIER();
    v8f acc0 = {}, acc1 = {};
    #pragma unroll
    for (int k4 = 0; k4 < 4; ++k4) {
      acc0 = wmma_bf16(a0[k4], b[k4], acc0);
      acc1 = wmma_bf16(a1[k4], b[k4], acc1);
    }
    const int col  = ct * 16 + n15;
    const float bias = b1[col];
    #pragma unroll
    for (int r = 0; r < 8; ++r) {
      const long t0 = (long)rt * 32 + r + grp * 8;
      ff1b[t0 * FDIM + col]        = (__bf16)fmaxf(acc0[r] + bias, 0.0f);
      ff1b[(t0 + 16) * FDIM + col] = (__bf16)fmaxf(acc1[r] + bias, 0.0f);
    }
  }
}

// out = LN2( relu(ff1 @ w2 + b2) + h )
__global__ void k_ffn2_ln2(const __bf16* __restrict__ ff1b, const __bf16* __restrict__ pW,
                           const float* __restrict__ b2, const float* __restrict__ hres,
                           const float* __restrict__ g2, const float* __restrict__ be2,
                           float* __restrict__ out) {
  __shared__ float tile[16 * DMODEL];
  const int rt   = blockIdx.x;
  const int wave = threadIdx.x >> 5;
  const int lane = threadIdx.x & 31;
  const int grp  = (lane >> 4) & 1;
  const int n15  = lane & 15;
  const long rowA = (long)rt * 16 + n15;

  v16bf a[12];                              // K = 384, loaded once per wave
  #pragma unroll
  for (int kk = 0; kk < 12; ++kk) a[kk] = load_a_frag(ff1b, FDIM, rowA, kk * 32);

  for (int nt = wave; nt < 8; nt += 4) {
    v8f acc = {};
    #pragma unroll
    for (int kg = 0; kg < 3; ++kg) {        // pipeline B in groups of 4
      v16bf b[4];
      #pragma unroll
      for (int k4 = 0; k4 < 4; ++k4) b[k4] = load_b_packed(pW, 8, kg * 4 + k4, nt);
      SCHED_BARRIER();
      #pragma unroll
      for (int k4 = 0; k4 < 4; ++k4) acc = wmma_bf16(a[kg * 4 + k4], b[k4], acc);
    }
    const int col  = nt * 16 + n15;
    const float bias = b2[col];
    #pragma unroll
    for (int r = 0; r < 8; ++r) {
      const int  rr = r + grp * 8;
      const long t  = (long)rt * 16 + rr;
      tile[rr * DMODEL + col] = fmaxf(acc[r] + bias, 0.0f) + hres[t * DMODEL + col];
    }
  }
  __syncthreads();
  if (threadIdx.x < 16) {
    const int  rr = threadIdx.x;
    const long t  = (long)rt * 16 + rr;
    float mean = 0.0f;
    for (int c = 0; c < DMODEL; ++c) mean += tile[rr * DMODEL + c];
    mean *= (1.0f / DMODEL);
    float var = 0.0f;
    for (int c = 0; c < DMODEL; ++c) {
      const float d = tile[rr * DMODEL + c] - mean; var += d * d;
    }
    var *= (1.0f / DMODEL);
    const float rstd = rsqrtf(var + LN_EPS);
    for (int c = 0; c < DMODEL; ++c)
      out[t * DMODEL + c] = g2[c] * (tile[rr * DMODEL + c] - mean) * rstd + be2[c];
  }
}

// ------------------------------------------------------------------ launch ---

extern "C" void kernel_launch(void* const* d_in, const int* in_sizes, int n_in,
                              void* d_out, int out_size, void* d_ws, size_t ws_size,
                              hipStream_t stream) {
  (void)in_sizes; (void)n_in; (void)out_size; (void)ws_size;
  const float* x     = (const float*)d_in[0];
  const int*   mask  = (const int*)d_in[1];
  const float* qkv_w = (const float*)d_in[2];
  const float* out_w = (const float*)d_in[3];
  const float* w1    = (const float*)d_in[4];
  const float* b1    = (const float*)d_in[5];
  const float* w2    = (const float*)d_in[6];
  const float* b2    = (const float*)d_in[7];
  const float* g1    = (const float*)d_in[8];
  const float* be1   = (const float*)d_in[9];
  const float* g2    = (const float*)d_in[10];
  const float* be2   = (const float*)d_in[11];

  char* ws = (char*)d_ws;                          // ~44.5 MB used
  __bf16* xb   = (__bf16*)(ws + ( 0ul << 20));     // 4 MB  x as bf16
  __bf16* qb   = (__bf16*)(ws + ( 4ul << 20));     // 4 MB  q [l,h,s,hd]
  __bf16* kbuf = (__bf16*)(ws + ( 8ul << 20));     // 4 MB  k [l,h,s,hd]
  __bf16* vtb  = (__bf16*)(ws + (12ul << 20));     // 4 MB  v^T [l,h,hd,s]
  __bf16* ctxb = (__bf16*)(ws + (16ul << 20));     // 4 MB  ctx [t,d]
  float*  hres = (float*) (ws + (20ul << 20));     // 8 MB  h (f32 residual)
  __bf16* hb   = (__bf16*)(ws + (28ul << 20));     // 4 MB  h as bf16
  __bf16* ff1b = (__bf16*)(ws + (32ul << 20));     // 12 MB ff1 [t,384]
  __bf16* pqkv = (__bf16*)(ws + (44ul << 20));                 // 96 KB packed
  __bf16* pout = (__bf16*)(ws + (44ul << 20) + (128ul << 10)); // 32 KB packed
  __bf16* pw1  = (__bf16*)(ws + (44ul << 20) + (256ul << 10)); // 96 KB packed
  __bf16* pw2  = (__bf16*)(ws + (44ul << 20) + (384ul << 10)); // 96 KB packed
  float*  out  = (float*)d_out;

  k_cvt_bf16 <<<512, 256, 0, stream>>>(x, xb, (NTOK * DMODEL) / 4);
  k_pack_b   <<<(4  * 24 * 32 + 255) / 256, 256, 0, stream>>>(qkv_w, pqkv, DMODEL, 3 * DMODEL);
  k_pack_b   <<<(4  *  8 * 32 + 255) / 256, 256, 0, stream>>>(out_w, pout, DMODEL, DMODEL);
  k_pack_b   <<<(4  * 24 * 32 + 255) / 256, 256, 0, stream>>>(w1,    pw1,  DMODEL, FDIM);
  k_pack_b   <<<(12 *  8 * 32 + 255) / 256, 256, 0, stream>>>(w2,    pw2,  FDIM,   DMODEL);

  k_qkv      <<<NTOK / 32, 128, 0, stream>>>(xb, pqkv, qb, kbuf, vtb);
  k_attn     <<<LNUM * NHEAD * (SEQ / 64), 128, 0, stream>>>(qb, kbuf, vtb, mask, ctxb);
  k_proj_ln1 <<<NTOK / 16, 128, 0, stream>>>(ctxb, pout, x, g1, be1, hres, hb);
  k_ffn1     <<<NTOK / 32, 128, 0, stream>>>(hb, pw1, b1, ff1b);
  k_ffn2_ln2 <<<NTOK / 16, 128, 0, stream>>>(ff1b, pw2, b2, hres, g2, be2, out);
}
